// EncoderS_44813688766728
// MI455X (gfx1250) — matrix operand order repaired and checked by reading, not compiled, not run on hardware
//
#include <hip/hip_runtime.h>
#include <cstddef>

// ---------------- CDNA5 / gfx1250 WMMA types ----------------
typedef __attribute__((ext_vector_type(16))) _Float16 v16h;
typedef __attribute__((ext_vector_type(8)))  _Float16 v8h;
typedef __attribute__((ext_vector_type(4)))  _Float16 v4h;
typedef __attribute__((ext_vector_type(8)))  float    v8f;
typedef __attribute__((ext_vector_type(4)))  float    v4f;

static __device__ __forceinline__ v16h cat16(v8h lo, v8h hi) {
    return __builtin_shufflevector(lo, hi, 0, 1, 2, 3, 4, 5, 6, 7,
                                           8, 9, 10, 11, 12, 13, 14, 15);
}
static __device__ __forceinline__ v4h cvt4(v4f v) {
    v4h h;
    h[0] = (_Float16)v[0]; h[1] = (_Float16)v[1];
    h[2] = (_Float16)v[2]; h[3] = (_Float16)v[3];
    return h;
}

// ---------------- Problem constants ----------------
constexpr int B_   = 16;
constexpr int S_   = 256;
constexpr int DIM_ = 41;
constexpr int ND_  = 42;    // DIM + 1 (time2vec channel)
constexpr int NK_  = 128;
constexpr int NH_  = 4;
constexpr int DH_  = 32;    // NK/NH
constexpr int NL_  = 3;

// =====================================================================
// 1. Ragged compaction: stable argsort(-mask) == ones first (in order),
//    then zeros (in order). One block (256 thr) per batch, scan in LDS.
// =====================================================================
__global__ __launch_bounds__(256)
void compact_kernel(const float* __restrict__ mask, int L,
                    int* __restrict__ order, int* __restrict__ totals)
{
    const int b   = blockIdx.x;
    const int tid = threadIdx.x;
    const int P   = S_ * ND_;            // 10752
    const int CH  = (P + 255) / 256;     // 42
    const int p0  = tid * CH;

    int cntl = 0;
    for (int j = 0; j < CH; ++j) {
        int p = p0 + j;
        if (p < P) {
            int c = p % ND_;
            int s = p / ND_;
            float mv = (c < DIM_) ? mask[((long)b * S_ + s) * DIM_ + c] : 1.0f;
            cntl += (mv != 0.0f) ? 1 : 0;
        }
    }
    __shared__ int sc[256];
    sc[tid] = cntl;
    __syncthreads();
    // Hillis-Steele inclusive scan
    for (int off = 1; off < 256; off <<= 1) {
        int v = (tid >= off) ? sc[tid - off] : 0;
        __syncthreads();
        sc[tid] += v;
        __syncthreads();
    }
    const int total       = sc[255];
    const int ones_before = sc[tid] - cntl;   // exclusive prefix
    if (tid == 0) totals[b] = total;

    int run = ones_before;
    for (int j = 0; j < CH; ++j) {
        int p = p0 + j;
        if (p >= P) break;
        int c = p % ND_;
        int s = p / ND_;
        float mv = (c < DIM_) ? mask[((long)b * S_ + s) * DIM_ + c] : 1.0f;
        int pos;
        if (mv != 0.0f) { pos = run; ++run; }
        else            { pos = total + (p - run); }   // zeros keep order
        if (pos < L) order[(long)b * L + pos] = p;
    }
}

// =====================================================================
// 2. Derive compacted per-edge scalars + target outputs
// =====================================================================
__global__ __launch_bounds__(256)
void derive_kernel(const float* __restrict__ ctx,  const float* __restrict__ value,
                   const float* __restrict__ tval, const float* __restrict__ tmask,
                   const float* __restrict__ w0,   const float* __restrict__ b0,
                   const int* __restrict__ order,  const int* __restrict__ totals,
                   int L,
                   float* __restrict__ mk,  float* __restrict__ Uval,
                   float* __restrict__ Uind, int* __restrict__ Tinds,
                   int* __restrict__ Cinds,
                   float* __restrict__ out_tU, float* __restrict__ out_tM)
{
    long i = (long)blockIdx.x * blockDim.x + threadIdx.x;
    if (i >= (long)B_ * L) return;
    int b = (int)(i / L);
    int l = (int)(i % L);
    int p = order[i];
    int s = p / ND_;
    int c = p % ND_;
    float m  = (l < totals[b]) ? 1.0f : 0.0f;
    float te = w0[0] * ctx[(long)b * S_ + s] + b0[0];
    float vf = (c < DIM_) ? value[((long)b * S_ + s) * DIM_ + c] : te;
    float tv = (c < DIM_) ? tval [((long)b * S_ + s) * DIM_ + c] : te;
    float tm = (c < DIM_) ? tmask[((long)b * S_ + s) * DIM_ + c] : 0.0f;
    float tUm = tv * m, tMm = tm * m;
    mk[i]     = m;
    out_tU[i] = tUm;
    out_tM[i] = tMm;
    Uval[i]   = vf * m;
    Uind[i]   = 1.0f - m + tMm;
    Tinds[i]  = (m != 0.0f) ? s : 0;
    Cinds[i]  = (m != 0.0f) ? c : 0;
}

// =====================================================================
// 3. Small init kernels (edge MLP K=2, Time2Vec sin, channel embed)
// =====================================================================
__global__ __launch_bounds__(256)
void init_U_kernel(const float* __restrict__ Uval, const float* __restrict__ Uind,
                   const float* __restrict__ mk,   const float* __restrict__ ew,
                   const float* __restrict__ eb,   float* __restrict__ U, int L)
{
    long i = (long)blockIdx.x * blockDim.x + threadIdx.x;
    if (i >= (long)B_ * L * NK_) return;
    int  n  = (int)(i % NK_);
    long bl = i / NK_;
    float v = Uval[bl] * ew[n] + Uind[bl] * ew[NK_ + n] + eb[n];
    U[i] = fmaxf(v, 0.0f) * mk[bl];
}

__global__ __launch_bounds__(256)
void init_T_kernel(const float* __restrict__ ctx, const float* __restrict__ tw,
                   const float* __restrict__ tb,  float* __restrict__ T)
{
    long i = (long)blockIdx.x * blockDim.x + threadIdx.x;
    if (i >= (long)B_ * S_ * NK_) return;
    int  n  = (int)(i % NK_);
    long bs = i / NK_;
    T[i] = sinf(ctx[bs] * tw[n] + tb[n]);
}

__global__ __launch_bounds__(256)
void init_C_kernel(const float* __restrict__ cw, const float* __restrict__ cb,
                   float* __restrict__ C)
{
    long i = (long)blockIdx.x * blockDim.x + threadIdx.x;
    if (i >= (long)B_ * ND_ * NK_) return;
    int  n  = (int)(i % NK_);
    long bc = i / NK_;
    int  c  = (int)(bc % ND_);
    C[i] = fmaxf(cw[(long)c * NK_ + n] + cb[n], 0.0f);
}

// =====================================================================
// 4. Gather / concat helpers
// =====================================================================
__global__ __launch_bounds__(256)
void gather_rows(const float* __restrict__ src, int srows,
                 const int* __restrict__ inds, float* __restrict__ dst,
                 int ldd, int off, int L)
{
    long i = (long)blockIdx.x * blockDim.x + threadIdx.x;
    if (i >= (long)B_ * L * NK_) return;
    int  n  = (int)(i % NK_);
    long bl = i / NK_;
    int  l  = (int)(bl % L);
    int  b  = (int)(bl / L);
    int  r  = inds[(long)b * L + l];
    dst[bl * ldd + off + n] = src[((long)b * srows + r) * NK_ + n];
}

__global__ __launch_bounds__(256)
void copy_cols(const float* __restrict__ src, float* __restrict__ dst,
               int ldd, int off, int L)
{
    long i = (long)blockIdx.x * blockDim.x + threadIdx.x;
    if (i >= (long)B_ * L * NK_) return;
    int  n  = (int)(i % NK_);
    long bl = i / NK_;
    dst[bl * ldd + off + n] = src[bl * NK_ + n];
}

// =====================================================================
// 5. WMMA GEMM workhorse: C = epilogue(A[M,K] @ W[K,N] + bias)
//    256 thr = 8 waves. Tile 128x128, K-step 32; each wave owns a
//    32x64 sub-tile -> 8 WMMAs per K-step per wave.
//    REQUIRES: N multiple of 128 (all calls use N=128), K multiple of 32.
//    Staging is two-phase: issue ALL float4 global loads first (8 in
//    flight), then convert+store, so loadcnt waits amortize.
//    ep: 0 -> acc+bias
//        1 -> R + relu(acc+bias)            (MAB output fc, residual)
//        2 -> relu(R + acc + bias)*mask[m]  (edge-update MLP)
// =====================================================================
__global__ __launch_bounds__(256)
void gemm_ep(const float* __restrict__ A, const float* __restrict__ W,
             const float* __restrict__ bias, const float* __restrict__ R,
             const float* __restrict__ rowmask, float* __restrict__ C,
             int M, int N, int K, int ep)
{
    __shared__ __align__(16) _Float16 As[128][40];
    __shared__ __align__(16) _Float16 BsT[128][40];

    const int m0   = blockIdx.y * 128;
    const int n0   = blockIdx.x * 128;
    const int tid  = threadIdx.x;
    const int lane = tid & 31;
    const int wv   = tid >> 5;
    const int mw   = (wv & 3) * 32;   // wave's m offset inside tile
    const int nw   = (wv >> 2) * 64;  // wave's n offset inside tile

    v8f acc[2][4];
#pragma unroll
    for (int mt = 0; mt < 2; ++mt)
#pragma unroll
        for (int t = 0; t < 4; ++t)
#pragma unroll
            for (int j = 0; j < 8; ++j) acc[mt][t][j] = 0.0f;

    const int arow = mw + (lane & 15);
    const int kb   = (lane >> 4) * 8;    // A half-offset: 0 or 8
    const int kb2  = (lane >> 4) * 8;    // B half-offset: 0 or 8
    const int ncl  = lane & 15;

    // Per-thread staging coordinates (fixed across K-steps)
    const int ar = tid >> 3;             // A row 0..31 (+32*ch)
    const int ac = (tid & 7) * 4;        // A col group
    const int br = tid >> 5;             // B k-row 0..7 (+8*ch)
    const int bc = (tid & 31) * 4;       // B n col group

    for (int k0 = 0; k0 < K; k0 += 32) {
        // prefetch next K-step tiles into L2 (global_prefetch_b8)
        if (k0 + 32 < K) {
            int gm = m0 + (tid >> 1);
            if (gm < M) __builtin_prefetch(A + (long)gm * K + k0 + 32 + (tid & 1) * 16, 0, 1);
            __builtin_prefetch(W + (long)(k0 + 32 + (tid >> 3)) * N + n0 + (tid & 7) * 16, 0, 1);
        }
        // ---- phase 1: issue all global loads (8 outstanding b128s) ----
        v4f av[4], bv4[4];
#pragma unroll
        for (int ch = 0; ch < 4; ++ch) {
            int r  = ar + ch * 32;
            int gm = m0 + r;
            v4f z = {0.0f, 0.0f, 0.0f, 0.0f};
            av[ch] = (gm < M) ? *(const v4f*)(A + (long)gm * K + k0 + ac) : z;
        }
#pragma unroll
        for (int ch = 0; ch < 4; ++ch) {
            int r = br + ch * 8;              // k index 0..31
            bv4[ch] = *(const v4f*)(W + (long)(k0 + r) * N + n0 + bc);
        }
        // ---- phase 2: convert + store to LDS ----
#pragma unroll
        for (int ch = 0; ch < 4; ++ch) {
            int r = ar + ch * 32;
            *(v4h*)&As[r][ac] = cvt4(av[ch]);
        }
#pragma unroll
        for (int ch = 0; ch < 4; ++ch) {
            int r = br + ch * 8;
            BsT[bc + 0][r] = (_Float16)bv4[ch][0];
            BsT[bc + 1][r] = (_Float16)bv4[ch][1];
            BsT[bc + 2][r] = (_Float16)bv4[ch][2];
            BsT[bc + 3][r] = (_Float16)bv4[ch][3];
        }
        __syncthreads();

        // A fragments (two 16-row subtiles), each = two b128 LDS loads
        v16h a0 = cat16(*(const v8h*)&As[arow][kb],
                        *(const v8h*)&As[arow][16 + kb]);
        v16h a1 = cat16(*(const v8h*)&As[arow + 16][kb],
                        *(const v8h*)&As[arow + 16][16 + kb]);
#pragma unroll
        for (int t = 0; t < 4; ++t) {
            const int bcol = nw + t * 16 + ncl;
            v16h bf = cat16(*(const v8h*)&BsT[bcol][kb2],
                            *(const v8h*)&BsT[bcol][kb2 + 16]);
            acc[0][t] = __builtin_amdgcn_wmma_f32_16x16x32_f16(
                false, a0, false, bf, (short)0, acc[0][t], false, false);
            acc[1][t] = __builtin_amdgcn_wmma_f32_16x16x32_f16(
                false, a1, false, bf, (short)0, acc[1][t], false, false);
        }
        __syncthreads();
    }

    // epilogue (C layout: VGPR i -> M = i (+8 for upper half-lanes), N = lane&15)
#pragma unroll
    for (int mt = 0; mt < 2; ++mt) {
        const int mbase = m0 + mw + mt * 16 + ((lane >> 4) * 8);
#pragma unroll
        for (int t = 0; t < 4; ++t) {
            int gn = n0 + nw + t * 16 + ncl;
            float bv = bias ? bias[gn] : 0.0f;
#pragma unroll
            for (int i = 0; i < 8; ++i) {
                int gm = mbase + i;
                if (gm >= M) continue;
                float v = acc[mt][t][i] + bv;
                if (ep == 1)      v = R[(long)gm * N + gn] + fmaxf(v, 0.0f);
                else if (ep == 2) v = fmaxf(R[(long)gm * N + gn] + v, 0.0f) * rowmask[gm];
                C[(long)gm * N + gn] = v;
            }
        }
    }
}

// =====================================================================
// 6. Attention scores: att[b,h,q,k] = (q_h . k_h) / sqrt(NK)
//    One WMMA (K=32) per 16x16 tile, vector loads straight from global.
// =====================================================================
__global__ __launch_bounds__(128)
void attn_scores(const float* __restrict__ q, const float* __restrict__ k,
                 float* __restrict__ att, int nq, int nk, float scale)
{
    const int bh   = blockIdx.z;
    const int b    = bh / NH_;
    const int h    = bh % NH_;
    const int q0   = blockIdx.y * 16;
    const int lane = threadIdx.x & 31;
    const int wv   = threadIdx.x >> 5;
    const int n0   = blockIdx.x * 256 + wv * 64;

    const float* qb = q + ((long)b * nq) * NK_ + h * DH_;
    const float* kp = k + ((long)b * nk) * NK_ + h * DH_;

    // A fragment: q rows (4x float4 loads, all 16B aligned)
    const int mrow = q0 + (lane & 15);
    const int kbo  = (lane >> 4) * 8;
    v16h a;
#pragma unroll
    for (int j = 0; j < 16; ++j) a[j] = (_Float16)0.0f;
    if (mrow < nq) {
        const float* ap = qb + (long)mrow * NK_;
        v4f f0 = *(const v4f*)(ap + kbo);
        v4f f1 = *(const v4f*)(ap + kbo + 4);
        v4f f2 = *(const v4f*)(ap + 16 + kbo);
        v4f f3 = *(const v4f*)(ap + 16 + kbo + 4);
#pragma unroll
        for (int j = 0; j < 4; ++j) {
            a[j]      = (_Float16)f0[j];
            a[j + 4]  = (_Float16)f1[j];
            a[j + 8]  = (_Float16)f2[j];
            a[j + 12] = (_Float16)f3[j];
        }
    }
    const int kb2  = (lane >> 4) * 8;
    const int moff = (lane >> 4) * 8;
#pragma unroll
    for (int t = 0; t < 4; ++t) {
        int key = n0 + t * 16 + (lane & 15);
        v16h bf;
#pragma unroll
        for (int j = 0; j < 16; ++j) bf[j] = (_Float16)0.0f;
        if (key < nk) {
            const float* bp = kp + (long)key * NK_ + kb2;
            v4f f0 = *(const v4f*)(bp + 0);
            v4f f1 = *(const v4f*)(bp + 4);
            v4f f2 = *(const v4f*)(bp + 16);
            v4f f3 = *(const v4f*)(bp + 20);
#pragma unroll
            for (int j = 0; j < 4; ++j) {
                bf[j]      = (_Float16)f0[j];
                bf[j + 4]  = (_Float16)f1[j];
                bf[j + 8]  = (_Float16)f2[j];
                bf[j + 12] = (_Float16)f3[j];
            }
        }
        v8f acc;
#pragma unroll
        for (int j = 0; j < 8; ++j) acc[j] = 0.0f;
        acc = __builtin_amdgcn_wmma_f32_16x16x32_f16(
            false, a, false, bf, (short)0, acc, false, false);
        if (key < nk) {
#pragma unroll
            for (int i = 0; i < 8; ++i) {
                int gm = q0 + moff + i;
                if (gm < nq)
                    att[((long)bh * nq + gm) * nk + key] = acc[i] * scale;
            }
        }
    }
}

// =====================================================================
// 7. Masked softmax over keys (in place). inds==NULL -> unmasked.
// =====================================================================
__global__ __launch_bounds__(256)
void softmax_kernel(float* __restrict__ att, int nq, int nk,
                    const int* __restrict__ inds, const float* __restrict__ mkv)
{
    const int row = blockIdx.x;
    const int bh  = row / nq;
    const int qi  = row - bh * nq;
    const int b   = bh / NH_;
    float* p = att + (long)row * nk;
    __shared__ float red[256];
    const int tid = threadIdx.x;

    float mx = -3.0e38f;
    for (int k = tid; k < nk; k += 256) {
        float s = p[k];
        if (inds) {
            bool ok = (inds[(long)b * nk + k] == qi) && (mkv[(long)b * nk + k] != 0.0f);
            if (!ok) { s = -1.0e10f; p[k] = s; }
        }
        mx = fmaxf(mx, s);
    }
    red[tid] = mx; __syncthreads();
    for (int off = 128; off > 0; off >>= 1) {
        if (tid < off) red[tid] = fmaxf(red[tid], red[tid + off]);
        __syncthreads();
    }
    mx = red[0]; __syncthreads();

    float sum = 0.0f;
    for (int k = tid; k < nk; k += 256) {
        float e = expf(p[k] - mx);
        p[k] = e;
        sum += e;
    }
    red[tid] = sum; __syncthreads();
    for (int off = 128; off > 0; off >>= 1) {
        if (tid < off) red[tid] += red[tid + off];
        __syncthreads();
    }
    float inv = 1.0f / red[0];
    for (int k = tid; k < nk; k += 256) p[k] *= inv;
}

// =====================================================================
// 8. A@V with q residual: o[b,q,h*32+d] = qres + sum_k A[q,k]*v[k,d]
//    K-loop over keys (step 32), V staged transposed to LDS so B
//    fragments are contiguous b128 loads; 2 WMMAs per step per wave.
// =====================================================================
__global__ __launch_bounds__(128)
void attn_av(const float* __restrict__ att, const float* __restrict__ v,
             const float* __restrict__ qres, float* __restrict__ o,
             int nq, int nk)
{
    const int bh   = blockIdx.z;
    const int b    = bh / NH_;
    const int h    = bh % NH_;
    const int lane = threadIdx.x & 31;
    const int wv   = threadIdx.x >> 5;
    const int m0   = blockIdx.x * 64 + wv * 16;

    const float* arow = att + (long)bh * nq * nk;
    const float* vbp  = v + ((long)b * nk) * NK_ + h * DH_;

    __shared__ __align__(16) _Float16 BsT[32][40];  // [d][k]
    v8f acc0, acc1;
#pragma unroll
    for (int j = 0; j < 8; ++j) { acc0[j] = 0.0f; acc1[j] = 0.0f; }

    const int kb2  = (lane >> 4) * 8;
    const int dcol = lane & 15;
    const int vr   = threadIdx.x >> 3;        // V k-row 0..15 (+16*ch)
    const int vd   = (threadIdx.x & 7) * 4;   // V d col group

    for (int k0 = 0; k0 < nk; k0 += 32) {
        // stage V tile 32x32 transposed: both loads issued before stores
        v4f val[2];
#pragma unroll
        for (int ch = 0; ch < 2; ++ch) {
            int kk = vr + ch * 16;
            v4f z = {0.0f, 0.0f, 0.0f, 0.0f};
            val[ch] = (k0 + kk < nk) ? *(const v4f*)(vbp + (long)(k0 + kk) * NK_ + vd) : z;
        }
#pragma unroll
        for (int ch = 0; ch < 2; ++ch) {
            int kk = vr + ch * 16;
            BsT[vd + 0][kk] = (_Float16)val[ch][0];
            BsT[vd + 1][kk] = (_Float16)val[ch][1];
            BsT[vd + 2][kk] = (_Float16)val[ch][2];
            BsT[vd + 3][kk] = (_Float16)val[ch][3];
        }
        __syncthreads();

        const int mrow = m0 + (lane & 15);
        const int kbo  = (lane >> 4) * 8;
        v16h a;
#pragma unroll
        for (int j = 0; j < 16; ++j) a[j] = (_Float16)0.0f;
        if (mrow < nq) {
            const float* ap = arow + (long)mrow * nk + k0;
            if (k0 + 32 <= nk) {
#pragma unroll
                for (int j = 0; j < 8; ++j) {
                    a[j]     = (_Float16)ap[kbo + j];
                    a[j + 8] = (_Float16)ap[16 + kbo + j];
                }
            } else {
#pragma unroll
                for (int j = 0; j < 8; ++j) {
                    if (k0 + kbo + j < nk)      a[j]     = (_Float16)ap[kbo + j];
                    if (k0 + 16 + kbo + j < nk) a[j + 8] = (_Float16)ap[16 + kbo + j];
                }
            }
        }
        v16h b0 = cat16(*(const v8h*)&BsT[dcol][kb2],
                        *(const v8h*)&BsT[dcol][kb2 + 16]);
        v16h b1 = cat16(*(const v8h*)&BsT[16 + dcol][kb2],
                        *(const v8h*)&BsT[16 + dcol][kb2 + 16]);
        acc0 = __builtin_amdgcn_wmma_f32_16x16x32_f16(
            false, a, false, b0, (short)0, acc0, false, false);
        acc1 = __builtin_amdgcn_wmma_f32_16x16x32_f16(
            false, a, false, b1, (short)0, acc1, false, false);
        __syncthreads();
    }

    const int moff = (lane >> 4) * 8;
#pragma unroll
    for (int i = 0; i < 8; ++i) {
        int gm = m0 + moff + i;
        if (gm < nq) {
            long base = ((long)b * nq + gm) * NK_ + h * DH_;
            o[base + dcol]      = qres[base + dcol]      + acc0[i];
            o[base + 16 + dcol] = qres[base + 16 + dcol] + acc1[i];
        }
    }
}

// =====================================================================
// 9. Final projection [3NK] -> 1
// =====================================================================
__global__ __launch_bounds__(256)
void final_out_kernel(const float* __restrict__ cat3, const float* __restrict__ w,
                      const float* __restrict__ bterm, float* __restrict__ out, long n)
{
    long i = (long)blockIdx.x * blockDim.x + threadIdx.x;
    if (i >= n) return;
    const float* r = cat3 + i * (3 * NK_);
    float s = bterm[0];
#pragma unroll 4
    for (int j = 0; j < 3 * NK_; ++j) s += r[j] * w[j];
    out[i] = s;
}

// =====================================================================
// Host orchestration
// =====================================================================
extern "C" void kernel_launch(void* const* d_in, const int* in_sizes, int n_in,
                              void* d_out, int out_size, void* d_ws, size_t ws_size,
                              hipStream_t stream)
{
    const int L = out_size / (3 * B_);

    auto f = [&](int i) { return (const float*)d_in[i]; };
    const float* ctx   = f(0);
    const float* value = f(1);
    const float* maskp = f(2);
    const float* tval  = f(3);
    const float* tmsk  = f(4);
    const float* t2w0  = f(5);
    const float* t2b0  = f(6);
    const float* ew    = f(7);
    const float* eb    = f(8);
    const float* cw    = f(9);
    const float* cb    = f(10);
    const float* tw    = f(11);
    const float* tb    = f(12);
    struct MW { const float *wq,*bq,*wk,*bk,*wv,*bv,*wo,*bo; int dk; };
    MW ct{f(13), f(14), f(15), f(16), f(17), f(18), f(19), f(20), 2 * NK_};
    MW tc{f(21), f(22), f(23), f(24), f(25), f(26), f(27), f(28), 2 * NK_};
    MW ca{f(29), f(30), f(31), f(32), f(33), f(34), f(35), f(36), NK_};
    const float* enw  = f(37);
    const float* enb  = f(38);
    const float* outw = f(39);
    const float* outb = f(40);

    float* out0 = (float*)d_out;
    float* tU   = out0 + (long)B_ * L;
    float* tM   = tU   + (long)B_ * L;

    // ---- workspace carve ----
    char*  base = (char*)d_ws;
    size_t off  = 0;
    auto alloc = [&](size_t bytes) -> void* {
        void* p = base + off;
        off = (off + bytes + 255) & ~(size_t)255;
        return p;
    };
    const long BL   = (long)B_ * L;
    const long BLN  = BL * NK_;
    const long BSN  = (long)B_ * S_ * NK_;
    const long BDN  = (long)B_ * ND_ * NK_;

    int*   totals = (int*)  alloc(B_ * sizeof(int));
    int*   order  = (int*)  alloc(BL * sizeof(int));
    int*   Tinds  = (int*)  alloc(BL * sizeof(int));
    int*   Cinds  = (int*)  alloc(BL * sizeof(int));
    float* mk     = (float*)alloc(BL * sizeof(float));
    float* Uval   = (float*)alloc(BL * sizeof(float));
    float* Uind   = (float*)alloc(BL * sizeof(float));
    float* U0     = (float*)alloc(BLN * sizeof(float));
    float* U1     = (float*)alloc(BLN * sizeof(float));
    float* T0     = (float*)alloc(BSN * sizeof(float));
    float* T1     = (float*)alloc(BSN * sizeof(float));
    float* Cb     = (float*)alloc(BDN * sizeof(float));
    float* C2     = (float*)alloc(BDN * sizeof(float));
    float* cat2   = (float*)alloc(BL * 2 * NK_ * sizeof(float));
    float* cat3   = (float*)alloc(BL * 3 * NK_ * sizeof(float));
    float* qbuf   = (float*)alloc(BSN * sizeof(float));
    float* kbuf   = (float*)alloc(BLN * sizeof(float));
    float* vbuf   = (float*)alloc(BLN * sizeof(float));
    float* obuf   = (float*)alloc(BSN * sizeof(float));
    float* att    = (float*)alloc((long)B_ * NH_ * S_ * L * sizeof(float));
    (void)ws_size; (void)n_in; (void)in_sizes;

    // ---- helpers ----
    auto gemm = [&](const float* A, const float* W, const float* bias,
                    const float* R, const float* rmask, float* Cc,
                    int M, int N, int K, int ep) {
        dim3 g((N + 127) / 128, (M + 127) / 128);
        gemm_ep<<<g, 256, 0, stream>>>(A, W, bias, R, rmask, Cc, M, N, K, ep);
    };
    auto gatherK = [&](const float* src, int srows, const int* inds,
                       float* dst, int ldd, int coff) {
        long n = BLN;
        gather_rows<<<(unsigned)((n + 255) / 256), 256, 0, stream>>>(
            src, srows, inds, dst, ldd, coff, L);
    };
    auto copyU = [&](const float* src, float* dst, int ldd, int coff) {
        long n = BLN;
        copy_cols<<<(unsigned)((n + 255) / 256), 256, 0, stream>>>(
            src, dst, ldd, coff, L);
    };
    const float scale = 0.088388347648318447f; // 1/sqrt(128)
    auto mab2 = [&](const float* Q, int nq, const float* Kin, int nk,
                    const MW& w, int layer, const int* inds, float* Oout) {
        int Mq = B_ * nq, Mk = B_ * nk;
        gemm(Q,   w.wq + (long)layer * NK_ * NK_,   w.bq + layer * NK_,
             nullptr, nullptr, qbuf, Mq, NK_, NK_, 0);
        gemm(Kin, w.wk + (long)layer * w.dk * NK_,  w.bk + layer * NK_,
             nullptr, nullptr, kbuf, Mk, NK_, w.dk, 0);
        gemm(Kin, w.wv + (long)layer * w.dk * NK_,  w.bv + layer * NK_,
             nullptr, nullptr, vbuf, Mk, NK_, w.dk, 0);
        dim3 gs((nk + 255) / 256, (nq + 15) / 16, B_ * NH_);
        attn_scores<<<gs, 128, 0, stream>>>(qbuf, kbuf, att, nq, nk, scale);
        softmax_kernel<<<B_ * NH_ * nq, 256, 0, stream>>>(att, nq, nk, inds, mk);
        dim3 ga((nq + 63) / 64, 1, B_ * NH_);
        attn_av<<<ga, 128, 0, stream>>>(att, vbuf, qbuf, obuf, nq, nk);
        gemm(obuf, w.wo + (long)layer * NK_ * NK_, w.bo + layer * NK_,
             obuf, nullptr, Oout, Mq, NK_, NK_, 1);
    };

    // ---- preprocessing ----
    compact_kernel<<<B_, 256, 0, stream>>>(maskp, L, order, totals);
    derive_kernel<<<(unsigned)((BL + 255) / 256), 256, 0, stream>>>(
        ctx, value, tval, tmsk, t2w0, t2b0, order, totals, L,
        mk, Uval, Uind, Tinds, Cinds, tU, tM);
    init_U_kernel<<<(unsigned)((BLN + 255) / 256), 256, 0, stream>>>(
        Uval, Uind, mk, ew, eb, U0, L);
    init_T_kernel<<<(unsigned)((BSN + 255) / 256), 256, 0, stream>>>(ctx, tw, tb, T0);
    init_C_kernel<<<(unsigned)((BDN + 255) / 256), 256, 0, stream>>>(cw, cb, Cb);

    // ---- layers ----
    float *U = U0, *Un = U1, *T = T0, *Tn = T1;
    for (int i = 0; i < NL_; ++i) {
        // k_t = gather(T, T_inds); cat2 = [k_t, U]; cat3[NK:2NK] = k_t; cat3[0:NK] = U
        gatherK(T, S_, Tinds, cat2, 2 * NK_, 0);
        gatherK(T, S_, Tinds, cat3, 3 * NK_, NK_);
        copyU(U, cat2, 2 * NK_, NK_);
        copyU(U, cat3, 3 * NK_, 0);
        // C2 = MAB2(C_, [k_t, U]) with channel mask
        mab2(Cb, ND_, cat2, L, ct, i, Cinds, C2);
        // k_c = gather(C_, C_inds); cat2 = [k_c, U]; cat3[2NK:3NK] = k_c
        gatherK(Cb, ND_, Cinds, cat2, 2 * NK_, 0);
        gatherK(Cb, ND_, Cinds, cat3, 3 * NK_, 2 * NK_);
        copyU(U, cat2, 2 * NK_, NK_);
        // T2 = MAB2(T_, [k_c, U]) with time mask
        mab2(T, S_, cat2, L, tc, i, Tinds, Tn);
        // U = relu(U + cat3 @ en_w + en_b) * mk
        gemm(cat3, enw + (long)i * 3 * NK_ * NK_, enb + i * NK_,
             U, mk, Un, (int)BL, NK_, 3 * NK_, 2);
        // C_ = MAB2(C2, C2) unmasked
        mab2(C2, ND_, C2, ND_, ca, i, nullptr, Cb);
        { float* t = U; U = Un; Un = t; }
        { float* t = T; T = Tn; Tn = t; }
    }

    // ---- final head ----
    gatherK(T,  S_,  Tinds, cat3, 3 * NK_, NK_);
    gatherK(Cb, ND_, Cinds, cat3, 3 * NK_, 2 * NK_);
    copyU(U, cat3, 3 * NK_, 0);
    final_out_kernel<<<(unsigned)((BL + 255) / 256), 256, 0, stream>>>(
        cat3, outw, outb, out0, BL);
}
